// CausalSelfAttention_72722386256484
// MI455X (gfx1250) — matrix operand order repaired and checked by reading, not compiled, not run on hardware
//
#include <hip/hip_runtime.h>
#include <math.h>

// ---------------------------------------------------------------------------
// Causal self-attention forward for MI455X (gfx1250).
// bf16 WMMA pipeline; GEMM tiles fed by async global->LDS DMA (ASYNCcnt);
// attention K/V tiles fed by the Tensor Data Mover (TENSORcnt), one
// tensor_load_to_lds per 64x64 tile. All WMMA fragments are contiguous
// b128 LDS reads (weights pre-transposed; V stored [B,H,D,T]).
// B=4, T=2048, C=1024, H=16, D=64
// ---------------------------------------------------------------------------

typedef __attribute__((ext_vector_type(16))) __bf16        v16bf;
typedef __attribute__((ext_vector_type(8)))  float         v8f;
typedef __attribute__((ext_vector_type(8)))  unsigned int  v8u;
typedef __attribute__((ext_vector_type(4)))  unsigned int  u32x4;
typedef __attribute__((ext_vector_type(8)))  unsigned int  u32x8;

static constexpr int Bc = 4;
static constexpr int Tc = 2048;
static constexpr int Cc = 1024;
static constexpr int Hc = 16;
static constexpr int Dc = 64;

// -------- bf16 helpers (storage type: unsigned short, RNE rounding) --------
__device__ __forceinline__ unsigned short f32_to_bf16(float f) {
  unsigned int u = __builtin_bit_cast(unsigned int, f);
  unsigned int r = u + 0x7FFFu + ((u >> 16) & 1u);
  return (unsigned short)(r >> 16);
}

// Fragment index maps per CDNA5 ISA 7.12.2 (16-bit A 16x32; B mirrored; C/D f32)
__device__ __forceinline__ int a_kidx(int p, int half) {
  return 16 * (p >> 2) + 2 * (p & 3) + 8 * half;   // A: VGPR p holds K pair
}
__device__ __forceinline__ int b_kidx(int p, int half) {
  return 16 * (p >> 2) + 2 * (p & 3) + 8 * half;   // B: VGPR p holds K pair
}

// -------- CDNA5 async global->LDS copy (per-lane 16B), ASYNCcnt tracked -----
__device__ __forceinline__ void async_copy_b128(void* lds_ptr, const void* gptr) {
  // Generic pointer to __shared__ : low 32 bits == LDS byte offset (ISA 10.2).
  unsigned lds_off = (unsigned)(unsigned long long)lds_ptr;
  unsigned long long ga = (unsigned long long)gptr;
  asm volatile("global_load_async_to_lds_b128 %0, %1, off"
               :
               : "v"(lds_off), "v"(ga)
               : "memory");
}

__device__ __forceinline__ void wait_async0() {
#if __has_builtin(__builtin_amdgcn_s_wait_asynccnt)
  __builtin_amdgcn_s_wait_asynccnt(0);
#else
  asm volatile("s_wait_asynccnt 0x0" ::: "memory");
#endif
}

// -------- CDNA5 Tensor Data Mover: 2D tile -> LDS, TENSORcnt tracked --------
// D# per CDNA5 ISA. 2-group (2D) form: count=1,
// type=2, data_size=2B, no pad/iterate/gather. Tensor dims == tile dims
// (tiles are fully interior), row stride in elements.
__device__ __forceinline__ void tdm_load_2d_bf16(
    unsigned lds_off, const void* gptr,
    unsigned tile_d0, unsigned tile_d1, unsigned stride0_elems) {
  unsigned long long ga = (unsigned long long)gptr;
  u32x4 g0;
  g0[0] = 1u;                                   // count=1 (user descriptor)
  g0[1] = lds_off;                              // lds_addr (bytes)
  g0[2] = (unsigned)ga;                         // global_addr[31:0]
  g0[3] = (unsigned)(ga >> 32) | (2u << 30);    // global_addr[56:32] | type=2
  u32x8 g1;
  g1[0] = 1u << 16;                             // data_size=1 -> 2 bytes
  g1[1] = (tile_d0 & 0xFFFFu) << 16;            // tensor_dim0[15:0] @bit48
  g1[2] = (tile_d0 >> 16) | ((tile_d1 & 0xFFFFu) << 16);  // dim0 hi | dim1 lo
  g1[3] = (tile_d1 >> 16) | (tile_d0 << 16);    // dim1 hi | tile_dim0 @112
  g1[4] = tile_d1;                              // tile_dim1 @128; tile_dim2=0
  g1[5] = stride0_elems;                        // tensor_dim0_stride[31:0]
  g1[6] = 0;                                    // stride0[47:32]=0 | stride1 lo
  g1[7] = 0;                                    // stride1 hi
  asm volatile("tensor_load_to_lds %0, %1"
               :
               : "s"(g0), "s"(g1)
               : "memory");
}

__device__ __forceinline__ void wait_tensor0() {
#if __has_builtin(__builtin_amdgcn_s_wait_tensorcnt)
  __builtin_amdgcn_s_wait_tensorcnt(0);
#else
  asm volatile("s_wait_tensorcnt 0x0" ::: "memory");
#endif
}

// ---------------------------------------------------------------------------
// Kernel 1a: fp32 -> bf16 conversion (vectorized x4), layout preserved
// ---------------------------------------------------------------------------
__global__ __launch_bounds__(256) void cvt_f32_bf16_x4(
    const float* __restrict__ in, unsigned short* __restrict__ out, int n4) {
  int i = blockIdx.x * blockDim.x + threadIdx.x;
  if (i < n4) {
    float4 f = ((const float4*)in)[i];
    unsigned int lo = (unsigned int)f32_to_bf16(f.x) |
                      ((unsigned int)f32_to_bf16(f.y) << 16);
    unsigned int hi = (unsigned int)f32_to_bf16(f.z) |
                      ((unsigned int)f32_to_bf16(f.w) << 16);
    ((uint2*)out)[i] = make_uint2(lo, hi);
  }
}

// ---------------------------------------------------------------------------
// Kernel 1b: W[K][N] fp32 -> Wt[N][K] bf16 (tiled transpose, coalesced I/O)
// ---------------------------------------------------------------------------
__global__ __launch_bounds__(256) void transpose_cvt_f32_bf16(
    const float* __restrict__ W, unsigned short* __restrict__ Wt,
    int K, int N) {
  __shared__ unsigned short t[64][72];   // pad 64->72 to spread banks
  const int k0 = blockIdx.y * 64;
  const int n0 = blockIdx.x * 64;
  const int tid = threadIdx.x;
#pragma unroll
  for (int i = 0; i < 16; ++i) {
    int idx = tid + i * 256;
    int r = idx >> 6, c = idx & 63;      // r: k-local, c: n-local
    t[r][c] = f32_to_bf16(W[(size_t)(k0 + r) * N + n0 + c]);
  }
  __syncthreads();
#pragma unroll
  for (int i = 0; i < 16; ++i) {
    int idx = tid + i * 256;
    int r = idx >> 6, c = idx & 63;      // r: n-local, c: k-local
    Wt[(size_t)(n0 + r) * K + k0 + c] = t[c][r];
  }
}

// ---------------------------------------------------------------------------
// Kernel 2/4: bf16 GEMM  Out[M,N] = A[M,K] * Wt[N,K]^T + bias[N]
//   MODE 0: scatter to q/k bf16 [B,H,T,D] and v bf16 [B,H,D,T] (N = 3C)
//   MODE 1: fp32 row-major output
// Block 256 threads (8 waves), tile 128x128x32; wave tile 32x64 (2x4 wmma).
// Double-buffered LDS fed by async DMA; DMA of tile i+1 overlaps WMMA of i.
// ---------------------------------------------------------------------------
template <int MODE>
__global__ __launch_bounds__(256) void gemm_bf16_kernel(
    const unsigned short* __restrict__ A,    // [M][K]
    const unsigned short* __restrict__ Wt,   // [N][K]
    const float* __restrict__ bias,
    unsigned short* __restrict__ q_out,
    unsigned short* __restrict__ k_out,
    unsigned short* __restrict__ v_out,
    float* __restrict__ out,
    int M, int N, int K) {
  constexpr int BM = 128, BN = 128, BK = 32;
  __shared__ alignas(16) unsigned short As[2][BM][BK];
  __shared__ alignas(16) unsigned short BsT[2][BN][BK];  // n-major, K-contig

  const int tid  = threadIdx.x;
  const int lane = tid & 31;
  const int wid  = tid >> 5;
  const int half = lane >> 4;
  const int l16  = lane & 15;
  const int wm   = wid >> 1;   // 0..3 : wave row (32 rows each)
  const int wn   = wid & 1;    // 0..1 : wave col (64 cols each)
  const int m0   = blockIdx.y * BM;
  const int n0   = blockIdx.x * BN;

  // per-thread DMA chunk coordinates (fixed across tiles); same shape for A/B
  const int r0 = tid >> 2;
  const int c0 = (tid & 3) << 3;
  const int r1 = (tid + 256) >> 2;
  const int c1 = ((tid + 256) & 3) << 3;

  auto issue_tile = [&](int k0, int buf) {
    async_copy_b128(&As[buf][r0][c0],  &A[(size_t)(m0 + r0) * K + k0 + c0]);
    async_copy_b128(&As[buf][r1][c1],  &A[(size_t)(m0 + r1) * K + k0 + c1]);
    async_copy_b128(&BsT[buf][r0][c0], &Wt[(size_t)(n0 + r0) * K + k0 + c0]);
    async_copy_b128(&BsT[buf][r1][c1], &Wt[(size_t)(n0 + r1) * K + k0 + c1]);
  };

  const v8f zero = {};
  v8f acc[2][4];
#pragma unroll
  for (int i = 0; i < 2; ++i)
#pragma unroll
    for (int j = 0; j < 4; ++j) acc[i][j] = zero;

  const int nk = K / BK;
  issue_tile(0, 0);

  for (int it = 0; it < nk; ++it) {
    const int buf = it & 1;
    wait_async0();          // my DMA for tile `it` has landed in LDS
    __syncthreads();        // everyone's tile `it` landed; iter it-1 reads done
    if (it + 1 < nk) issue_tile((it + 1) * BK, buf ^ 1);  // overlap with WMMA

    // ---- hoist all fragment loads (contiguous b128 LDS reads) ----
    v16bf afrag[2];
#pragma unroll
    for (int i = 0; i < 2; ++i) {
      int row = wm * 32 + i * 16 + l16;
      v8u u;
#pragma unroll
      for (int p = 0; p < 8; ++p)
        u[p] = *(const unsigned int*)&As[buf][row][a_kidx(p, half)];
      afrag[i] = __builtin_bit_cast(v16bf, u);
    }
    v16bf bfrag[4];
#pragma unroll
    for (int j = 0; j < 4; ++j) {
      int col = wn * 64 + j * 16 + l16;
      v8u u;
#pragma unroll
      for (int p = 0; p < 8; ++p)
        u[p] = *(const unsigned int*)&BsT[buf][col][b_kidx(p, half)];
      bfrag[j] = __builtin_bit_cast(v16bf, u);
    }

    // ---- 8 WMMAs per wave per K-step ----
#pragma unroll
    for (int i = 0; i < 2; ++i)
#pragma unroll
      for (int j = 0; j < 4; ++j)
        acc[i][j] = __builtin_amdgcn_wmma_f32_16x16x32_bf16(
            false, afrag[i], false, bfrag[j], (short)0, acc[i][j], false, false);
  }

  // ---- epilogue: bias + store ----
#pragma unroll
  for (int i = 0; i < 2; ++i) {
#pragma unroll
    for (int j = 0; j < 4; ++j) {
      int col_l = wn * 64 + j * 16 + l16;
      int n = n0 + col_l;
      float bval = bias[n];
#pragma unroll
      for (int v = 0; v < 8; ++v) {
        int m = m0 + wm * 32 + i * 16 + v + 8 * half;
        float val = acc[i][j][v] + bval;
        if (MODE == 0) {
          int which = n >> 10;         // 0:q 1:k 2:v
          int c = n & 1023;
          int h = c >> 6;
          int d = c & 63;
          int b = m >> 11;             // / T (2048)
          int t = m & 2047;
          unsigned short bf = f32_to_bf16(val);
          size_t head = ((size_t)b * Hc + h) * (size_t)Tc * Dc;
          if (which == 0)      q_out[head + (size_t)t * Dc + d] = bf;
          else if (which == 1) k_out[head + (size_t)t * Dc + d] = bf;
          else                 v_out[head + (size_t)d * Tc + t] = bf;  // V^T
        } else {
          out[(size_t)m * N + n] = val;
        }
      }
    }
  }
}

// ---------------------------------------------------------------------------
// Kernel 3: flash attention. Grid (T/128, H, B), 256 threads (8 waves).
// Each wave owns 16 q rows; K tiles [KV][D] and V^T tiles [D][KV] are moved
// by the Tensor Data Mover (one tensor_load_to_lds per tile, issued by wave 0,
// fenced with s_wait_tensorcnt + workgroup barrier), double-buffered so the
// TDM DMA of tile i+1 overlaps the WMMA/softmax of tile i.
// ---------------------------------------------------------------------------
__global__ __launch_bounds__(256) void attn_kernel(
    const unsigned short* __restrict__ Q,   // [B,H,T,D] bf16
    const unsigned short* __restrict__ Kt,  // [B,H,T,D] bf16
    const unsigned short* __restrict__ Vt,  // [B,H,D,T] bf16 (transposed)
    unsigned short* __restrict__ Y) {       // [B,T,C]   bf16
  constexpr int BQ = 128, BKV = 64;
  __shared__ alignas(16) unsigned short Ks[2][BKV][Dc];   // [kv][d]
  __shared__ alignas(16) unsigned short VsT[2][Dc][BKV];  // [d][kv]
  __shared__ alignas(16) unsigned short Ps[8][16][BKV];   // per-wave P staging

  const int tid  = threadIdx.x;
  const int lane = tid & 31;
  const int wid  = tid >> 5;
  const int half = lane >> 4;
  const int l16  = lane & 15;
  const int b    = blockIdx.z;
  const int h    = blockIdx.y;
  const int q0   = blockIdx.x * BQ;
  const int qrow0 = q0 + wid * 16;

  const size_t head_off = ((size_t)b * Hc + h) * (size_t)Tc * Dc;
  const unsigned short* Qh = Q + head_off;
  const unsigned short* Kh = Kt + head_off;
  const unsigned short* Vh = Vt + head_off;   // [D][T] within head

  const float sc = 0.125f * 1.44269504088896340736f;  // 1/sqrt(D) * log2(e)

  // one TDM descriptor pair per KV tile, issued by wave 0 only
  auto issue_kv = [&](int kv0, int buf) {
    if (wid == 0) {
      tdm_load_2d_bf16((unsigned)(unsigned long long)&Ks[buf][0][0],
                       Kh + (size_t)kv0 * Dc, Dc, BKV, (unsigned)Dc);
      tdm_load_2d_bf16((unsigned)(unsigned long long)&VsT[buf][0][0],
                       Vh + kv0, BKV, Dc, (unsigned)Tc);
    }
  };

  // ---- Q fragments (kept in registers; 2 chunks of K=32 over D=64) ----
  v16bf qfrag[2];
#pragma unroll
  for (int kc = 0; kc < 2; ++kc) {
    v8u u;
    int row = qrow0 + l16;
#pragma unroll
    for (int p = 0; p < 8; ++p) {
      int kk = kc * 32 + a_kidx(p, half);
      u[p] = *(const unsigned int*)&Qh[(size_t)row * Dc + kk];
    }
    qfrag[kc] = __builtin_bit_cast(v16bf, u);
  }

  const v8f zero = {};
  float m_r[8], l_r[8];
  v8f o_acc[4];
#pragma unroll
  for (int v = 0; v < 8; ++v) { m_r[v] = -1e30f; l_r[v] = 0.f; }
#pragma unroll
  for (int j = 0; j < 4; ++j) o_acc[j] = zero;

  const int n_tiles = (q0 + BQ) / BKV;  // causal bound for this block
  issue_kv(0, 0);

  for (int it = 0; it < n_tiles; ++it) {
    const int kv0 = it * BKV;
    const int buf = it & 1;
    if (wid == 0) wait_tensor0();   // wave 0's TDM for tile `it` done
    __syncthreads();                // release all waves; prev reads retired
    if (it + 1 < n_tiles) issue_kv(kv0 + BKV, buf ^ 1);  // overlap with WMMA

    // ---- hoist all K fragments, then S = Q*K^T WMMA burst ----
    v16bf kf[4][2];
#pragma unroll
    for (int j = 0; j < 4; ++j) {
      int kvl = j * 16 + l16;            // column of S == kv row
#pragma unroll
      for (int kc = 0; kc < 2; ++kc) {
        v8u u;
#pragma unroll
        for (int p = 0; p < 8; ++p)
          u[p] = *(const unsigned int*)&Ks[buf][kvl][kc * 32 + b_kidx(p, half)];
        kf[j][kc] = __builtin_bit_cast(v16bf, u);
      }
    }
    v8f s[4];
#pragma unroll
    for (int j = 0; j < 4; ++j) {
      v8f a = __builtin_amdgcn_wmma_f32_16x16x32_bf16(
          false, qfrag[0], false, kf[j][0], (short)0, zero, false, false);
      s[j] = __builtin_amdgcn_wmma_f32_16x16x32_bf16(
          false, qfrag[1], false, kf[j][1], (short)0, a, false, false);
    }

    // ---- scale + causal mask ----
#pragma unroll
    for (int j = 0; j < 4; ++j) {
      int kg = kv0 + j * 16 + l16;
#pragma unroll
      for (int v = 0; v < 8; ++v) {
        int qg = qrow0 + v + 8 * half;
        float val = s[j][v] * sc;
        s[j][v] = (kg <= qg) ? val : -1e30f;
      }
    }

    // ---- row max (half-wave butterfly over 16 lanes) ----
    float alpha[8];
#pragma unroll
    for (int v = 0; v < 8; ++v) {
      float mx = fmaxf(fmaxf(s[0][v], s[1][v]), fmaxf(s[2][v], s[3][v]));
#pragma unroll
      for (int off = 1; off < 16; off <<= 1)
        mx = fmaxf(mx, __shfl_xor(mx, off, 32));
      float mn = fmaxf(m_r[v], mx);
      alpha[v] = exp2f(m_r[v] - mn);
      m_r[v] = mn;
    }

    // ---- P = exp2(S - m), row sums, rescale accumulators ----
    float rs[8];
#pragma unroll
    for (int v = 0; v < 8; ++v) rs[v] = 0.f;
#pragma unroll
    for (int j = 0; j < 4; ++j) {
#pragma unroll
      for (int v = 0; v < 8; ++v) {
        float p = exp2f(s[j][v] - m_r[v]);
        s[j][v] = p;
        rs[v] += p;
      }
    }
#pragma unroll
    for (int v = 0; v < 8; ++v) {
      float t = rs[v];
#pragma unroll
      for (int off = 1; off < 16; off <<= 1) t += __shfl_xor(t, off, 32);
      l_r[v] = l_r[v] * alpha[v] + t;
#pragma unroll
      for (int j = 0; j < 4; ++j) o_acc[j][v] *= alpha[v];
    }

    // ---- stage P (C/D layout) -> LDS -> reload in A layout ----
#pragma unroll
    for (int j = 0; j < 4; ++j)
#pragma unroll
      for (int v = 0; v < 8; ++v)
        Ps[wid][v + 8 * half][j * 16 + l16] = f32_to_bf16(s[j][v]);

    // ---- hoist V fragments (contiguous: VsT is [d][kv]) and P A-frags ----
    v16bf vf[2][4];
#pragma unroll
    for (int kc = 0; kc < 2; ++kc) {
#pragma unroll
      for (int j = 0; j < 4; ++j) {
        int col = j * 16 + l16;          // d column
        v8u u;
#pragma unroll
        for (int p = 0; p < 8; ++p)
          u[p] = *(const unsigned int*)&VsT[buf][col][kc * 32 + b_kidx(p, half)];
        vf[kc][j] = __builtin_bit_cast(v16bf, u);
      }
    }
    v16bf pa[2];
#pragma unroll
    for (int kc = 0; kc < 2; ++kc) {
      v8u u;
#pragma unroll
      for (int p = 0; p < 8; ++p)
        u[p] = *(const unsigned int*)&Ps[wid][l16][kc * 32 + a_kidx(p, half)];
      pa[kc] = __builtin_bit_cast(v16bf, u);
    }

    // ---- O += P * V WMMA burst ----
#pragma unroll
    for (int kc = 0; kc < 2; ++kc)
#pragma unroll
      for (int j = 0; j < 4; ++j)
        o_acc[j] = __builtin_amdgcn_wmma_f32_16x16x32_bf16(
            false, pa[kc], false, vf[kc][j], (short)0, o_acc[j], false, false);
  }

  // ---- normalize and write y (bf16, [B,T,C] with col = h*D + d) ----
#pragma unroll
  for (int j = 0; j < 4; ++j) {
#pragma unroll
    for (int v = 0; v < 8; ++v) {
      int qg = qrow0 + v + 8 * half;
      int d = j * 16 + l16;
      float val = o_acc[j][v] / l_r[v];
      Y[((size_t)b * Tc + qg) * Cc + h * Dc + d] = f32_to_bf16(val);
    }
  }
}

// ---------------------------------------------------------------------------
// Host launcher
// ---------------------------------------------------------------------------
extern "C" void kernel_launch(void* const* d_in, const int* in_sizes, int n_in,
                              void* d_out, int out_size, void* d_ws, size_t ws_size,
                              hipStream_t stream) {
  (void)in_sizes; (void)n_in; (void)out_size; (void)ws_size;
  const float* x     = (const float*)d_in[0];  // [B,T,C]
  const float* Wqkv  = (const float*)d_in[1];  // [C,3C]
  const float* bqkv  = (const float*)d_in[2];  // [3C]
  const float* Wproj = (const float*)d_in[3];  // [C,C]
  const float* bproj = (const float*)d_in[4];  // [C]
  float* out = (float*)d_out;                  // [B,T,C] fp32

  const size_t M = (size_t)Bc * Tc;            // 8192
  char* ws = (char*)d_ws;
  size_t off = 0;
  auto alloc = [&](size_t bytes) -> void* {
    void* p = ws + off;
    off += (bytes + 255) & ~(size_t)255;
    return p;
  };
  unsigned short* x_bf  = (unsigned short*)alloc(M * Cc * 2);
  unsigned short* wq_t  = (unsigned short*)alloc((size_t)3 * Cc * Cc * 2);  // [3C][C]
  unsigned short* wp_t  = (unsigned short*)alloc((size_t)Cc * Cc * 2);      // [C][C]
  unsigned short* q_bf  = (unsigned short*)alloc(M * Cc * 2);               // [B,H,T,D]
  unsigned short* k_bf  = (unsigned short*)alloc(M * Cc * 2);               // [B,H,T,D]
  unsigned short* v_bf  = (unsigned short*)alloc(M * Cc * 2);               // [B,H,D,T]
  unsigned short* y_bf  = (unsigned short*)alloc(M * Cc * 2);               // [B,T,C]

  // 1) fp32 -> bf16: x straight; weights transposed to K-contiguous
  {
    int n4 = (int)(M * Cc / 4);
    cvt_f32_bf16_x4<<<(n4 + 255) / 256, 256, 0, stream>>>(x, x_bf, n4);
    dim3 gq((3 * Cc) / 64, Cc / 64);
    transpose_cvt_f32_bf16<<<gq, 256, 0, stream>>>(Wqkv, wq_t, Cc, 3 * Cc);
    dim3 gp(Cc / 64, Cc / 64);
    transpose_cvt_f32_bf16<<<gp, 256, 0, stream>>>(Wproj, wp_t, Cc, Cc);
  }

  // 2) QKV projection GEMM (scatter q/k [B,H,T,D], v [B,H,D,T])
  {
    dim3 grid((3 * Cc) / 128, (unsigned)(M / 128));
    gemm_bf16_kernel<0><<<grid, 256, 0, stream>>>(
        x_bf, wq_t, bqkv, q_bf, k_bf, v_bf, nullptr,
        (int)M, 3 * Cc, Cc);
  }

  // 3) flash attention (TDM-fed K/V tiles)
  {
    dim3 grid(Tc / 128, Hc, Bc);
    attn_kernel<<<grid, 256, 0, stream>>>(q_bf, k_bf, v_bf, y_bf);
  }

  // 4) output projection GEMM -> fp32
  {
    dim3 grid(Cc / 128, (unsigned)(M / 128));
    gemm_bf16_kernel<1><<<grid, 256, 0, stream>>>(
        y_bf, wp_t, bproj, nullptr, nullptr, nullptr, out,
        (int)M, Cc, Cc);
  }
}
